// STFT_10548439679392
// MI455X (gfx1250) — matrix operand order repaired and checked by reading, not compiled, not run on hardware
//
#include <hip/hip_runtime.h>
#include <hip/hip_bf16.h>
#include <math.h>

typedef __attribute__((ext_vector_type(16))) _Float16 v16h;
typedef __attribute__((ext_vector_type(8)))  _Float16 v8h;
typedef __attribute__((ext_vector_type(8)))  float    v8f;

#define WINLEN   320
#define HOP      160
#define NSAMP    320000
#define TFRAMES  2002
#define BATCH    32
#define MROWS    (BATCH * TFRAMES)   /* 64064 */
#define MTILES   (MROWS / 16)        /* 4004  */
#define NTILES   (WINLEN / 16)       /* 20    */
#define KSTEPS   (WINLEN / 32)       /* 10    */
#define EPSF     1.1920928955078125e-07f

/* ws layout:
   rseg  : float[160]                       @ byte 0      (reciprocal norms)
   Bpack : _Float16[NTILES][KSTEPS][32][16] @ byte 4096   (pre-packed B frags, 200 KB) */

__global__ void setup_tables(void* __restrict__ ws, const float* __restrict__ win) {
    float*    rseg  = (float*)ws;
    _Float16* Bpack = (_Float16*)((char*)ws + 4096);
    __shared__ float gsh[WINLEN];

    const int t = threadIdx.x;                // 0..319
    const double PI2 = 6.283185307179586476925286766559;
    if (t < WINLEN) {
        // g[d] = (1/320) [ 1 + cos(pi d) + 2 sum_{k=1}^{159} cos(2 pi k d / 320) ]
        double acc = 0.0;
        for (int k = 0; k <= 160; ++k) {
            double w = (k == 0 || k == 160) ? 1.0 : 2.0;
            acc += w * cos(PI2 * (double)k * (double)t / 320.0);
        }
        gsh[t] = (float)(acc / 320.0);
    }
    if (t < HOP) {
        rseg[t] = 1.0f / (win[t] + win[t + HOP] + EPSF);
    }
    __syncthreads();

    // Bpack[nt][ks][lane][i] = g[(ks*32 + 16*(lane>>4) + i - (nt*16 + (lane&15))) mod 320]
    const int total = NTILES * KSTEPS * 32;   // 6400 lane-entries of 16 f16
    for (int e = t; e < total; e += blockDim.x) {
        int lane  = e & 31;
        int rest  = e >> 5;
        int ks    = rest % KSTEPS;
        int nt    = rest / KSTEPS;
        int n_g   = nt * 16 + (lane & 15);
        int kbase = ks * 32 + 16 * (lane >> 4);
        _Float16* dst = Bpack + (size_t)e * 16;
        for (int i = 0; i < 16; ++i) {
            int d = kbase + i - n_g;
            if (d < 0) d += WINLEN;            // d in (-320, 320)
            dst[i] = (_Float16)gsh[d];
        }
    }
}

__global__ void zero_out(float* __restrict__ p, long n) {
    long i = (long)blockIdx.x * blockDim.x + threadIdx.x;
    if (i < n) p[i] = 0.0f;
}

__global__ void __launch_bounds__(32)
stft_gemm_ola(const float* __restrict__ sig, const float* __restrict__ win,
              const void* __restrict__ ws, float* __restrict__ out) {
    __shared__ __align__(16) _Float16 Ah[16][WINLEN];  // hi f16 of windowed frames
    __shared__ __align__(16) _Float16 Al[16][WINLEN];  // lo f16 (wf - (f16)wf)
    __shared__ float rsegL[HOP];

    const float* rseg  = (const float*)ws;
    const v16h*  Bpack = (const v16h*)((const char*)ws + 4096);

    const int lane = threadIdx.x;
    const int mt   = blockIdx.x;

    for (int i = lane; i < HOP; i += 32) rsegL[i] = rseg[i];

    // Stage 16 rows; each lane handles chunks of 8 contiguous samples.
    for (int c = lane; c < 16 * 40; c += 32) {
        int row = c / 40;
        int m0  = (c - row * 40) * 8;
        int r   = mt * 16 + row;
        int b   = r / TFRAMES;
        int t   = r - b * TFRAMES;
        int sidx0 = t * HOP + m0 - HOP;                 // padded coord -> sig index
        const float* sp = sig + (long)b * NSAMP + sidx0;
        float x[8];
        if (sidx0 >= 0 && sidx0 + 8 <= NSAMP) {
            const float4* p4 = (const float4*)sp;       // 16B aligned (sidx0 % 8 == 0)
            float4 q0 = p4[0], q1 = p4[1];
            x[0]=q0.x; x[1]=q0.y; x[2]=q0.z; x[3]=q0.w;
            x[4]=q1.x; x[5]=q1.y; x[6]=q1.z; x[7]=q1.w;
        } else {
#pragma unroll
            for (int i = 0; i < 8; ++i) {
                int si = sidx0 + i;
                x[i] = (si >= 0 && si < NSAMP) ? sp[i] : 0.0f;
            }
        }
        v8h hv, lv;
#pragma unroll
        for (int i = 0; i < 8; ++i) {
            float wf = x[i] * win[m0 + i];
            _Float16 h = (_Float16)wf;
            hv[i] = h;
            lv[i] = (_Float16)(wf - (float)h);
        }
        *(v8h*)&Ah[row][m0] = hv;
        *(v8h*)&Al[row][m0] = lv;
    }
    __syncthreads();

    const int halfsel = lane >> 4;     // 0: lanes 0-15, 1: lanes 16-31
    const int rn      = lane & 15;     // A-row / B-col within tile
    const int off     = halfsel * 16;

    // Per-output-row constants (hoisted integer magic-divisions).
    int  base_s[8];
    long baseidx[8];
#pragma unroll
    for (int i = 0; i < 8; ++i) {
        int r = mt * 16 + i + 8 * halfsel;
        int b = r / TFRAMES;
        int t = r - b * TFRAMES;
        base_s[i]  = (t - 1) * HOP;
        baseidx[i] = (long)b * NSAMP;
    }

    for (int nt = 0; nt < NTILES; ++nt) {
        const int n_g = nt * 16 + rn;
        v8f acc_h = {};   // two independent WMMA accumulation chains
        v8f acc_l = {};
#pragma unroll
        for (int ks = 0; ks < KSTEPS; ++ks) {
            const int kb = ks * 32;
            v16h bf = Bpack[(nt * KSTEPS + ks) * 32 + lane];
            v8h a0 = *(const v8h*)&Ah[rn][kb + off];
            v8h a1 = *(const v8h*)&Ah[rn][kb + off + 8];
            v8h l0 = *(const v8h*)&Al[rn][kb + off];
            v8h l1 = *(const v8h*)&Al[rn][kb + off + 8];
            v16h ah = __builtin_shufflevector(a0, a1, 0,1,2,3,4,5,6,7,8,9,10,11,12,13,14,15);
            v16h al = __builtin_shufflevector(l0, l1, 0,1,2,3,4,5,6,7,8,9,10,11,12,13,14,15);
            acc_h = __builtin_amdgcn_wmma_f32_16x16x32_f16(
                        false, ah, false, bf, (short)0, acc_h, false, false);
            acc_l = __builtin_amdgcn_wmma_f32_16x16x32_f16(
                        false, al, false, bf, (short)0, acc_l, false, false);
        }
        // Fused overlap-add + normalization (multiply by precomputed reciprocal).
#pragma unroll
        for (int i = 0; i < 8; ++i) {
            int s = base_s[i] + n_g;
            if ((unsigned)s < (unsigned)NSAMP) {
                float v = (acc_h[i] + acc_l[i]) * rsegL[(unsigned)s / 2001u];
                atomicAdd(&out[baseidx[i] + s], v);
            }
        }
    }
}

extern "C" void kernel_launch(void* const* d_in, const int* in_sizes, int n_in,
                              void* d_out, int out_size, void* d_ws, size_t ws_size,
                              hipStream_t stream) {
    const float* sig = (const float*)d_in[0];
    const float* win = (const float*)d_in[1];
    float* out = (float*)d_out;

    setup_tables<<<1, WINLEN, 0, stream>>>(d_ws, win);

    long n = (long)out_size;
    int zgrid = (int)((n + 255) / 256);
    zero_out<<<zgrid, 256, 0, stream>>>(out, n);

    stft_gemm_ola<<<MTILES, 32, 0, stream>>>(sig, win, d_ws, out);
}
